// Llama4MoE_4896262718157
// MI455X (gfx1250) — matrix operand order repaired and checked by reading, not compiled, not run on hardware
//
#include <hip/hip_runtime.h>
#include <hip/hip_bf16.h>

// Problem constants (match reference)
#define T_TOK 512
#define H_DIM 2048
#define E_EXP 8
#define I_DIM 4096

#define TM 16      // M tile (rows per block)
#define TN 128     // N tile (8 waves * 16 columns)
#define BK 32      // K per WMMA (bf16 16x16x32)
#define BKT 64     // K per staged iteration (2 WMMA sub-steps)
#define PAD_ROWS 128

typedef __attribute__((ext_vector_type(16))) __bf16         v16bf;
typedef __attribute__((ext_vector_type(8)))  float          v8f;
typedef __attribute__((ext_vector_type(8)))  unsigned int   v8u;

// ---- fp32 -> bf16 (round half up) ----
__device__ __forceinline__ unsigned short f2bf(float f) {
  unsigned int u = __builtin_bit_cast(unsigned int, f);
  return (unsigned short)((u + 0x8000u) >> 16);
}

// ---- rounded packed pair: lo -> [15:0], hi -> [31:16] (activations) ----
__device__ __forceinline__ unsigned int pkbf(float lo, float hi) {
  unsigned int ulo = __builtin_bit_cast(unsigned int, lo) + 0x8000u;
  unsigned int uhi = __builtin_bit_cast(unsigned int, hi) + 0x8000u;
#if __has_builtin(__builtin_amdgcn_perm)
  return __builtin_amdgcn_perm(uhi, ulo, 0x07060302u);
#else
  return (uhi & 0xFFFF0000u) | (ulo >> 16);
#endif
}

// ---- truncating packed pair: 1 VALU per dword (weight stream) ----
__device__ __forceinline__ unsigned int pktr(unsigned int ulo, unsigned int uhi) {
#if __has_builtin(__builtin_amdgcn_perm)
  return __builtin_amdgcn_perm(uhi, ulo, 0x07060302u);
#else
  return (uhi & 0xFFFF0000u) | (ulo >> 16);
#endif
}

__device__ __forceinline__ float fast_rcp(float x) {
#if __has_builtin(__builtin_amdgcn_rcpf)
  return __builtin_amdgcn_rcpf(x);
#else
  return 1.0f / x;
#endif
}

// ---- B fragment: lane = column (lane&15), K contiguous in memory.
// Lanes 0-15 hold K=[0,16), lanes 16-31 K=[16,32) (kOff folded into ptr).
// 16 fp32 -> 8 packed bf16 dwords, one v_perm each (truncate).
__device__ __forceinline__ v16bf load_b_frag(const float* __restrict__ p) {
  const uint4* p4 = (const uint4*)p;
  uint4 v0 = p4[0], v1 = p4[1], v2 = p4[2], v3 = p4[3];
  v8u s;
  s[0] = pktr(v0.x, v0.y);
  s[1] = pktr(v0.z, v0.w);
  s[2] = pktr(v1.x, v1.y);
  s[3] = pktr(v1.z, v1.w);
  s[4] = pktr(v2.x, v2.y);
  s[5] = pktr(v2.z, v2.w);
  s[6] = pktr(v3.x, v3.y);
  s[7] = pktr(v3.z, v3.w);
  return __builtin_bit_cast(v16bf, s);
}

// ---- A fragment from LDS tile As[16][BKT] (bf16), sub-chunk sel in {0,1}.
// lane<16: row=lane,  K = 0..15 (VGPR0-7)
// lane>=16: row=lane-16, K = 16..31
__device__ __forceinline__ v16bf load_a_frag(const unsigned short* As, int lane, int sub) {
  int m  = lane & 15;
  int kb = (lane < 16) ? 0 : 16;
  const uint4* q = (const uint4*)(As + m * BKT + sub * BK + kb);  // 16B aligned
  uint4 lo = q[0];   // K = kb .. kb+7
  uint4 hi = q[1];   // K = kb+8 .. kb+15
  v8u s;
  s[0] = lo.x; s[1] = lo.y; s[2] = lo.z; s[3] = lo.w;
  s[4] = hi.x; s[5] = hi.y; s[6] = hi.z; s[7] = hi.w;
  return __builtin_bit_cast(v16bf, s);
}

// ============================================================
// 1) Router: top-1 argmax, sigmoid scale, atomic slot in expert bucket
// ============================================================
__global__ __launch_bounds__(256) void moe_router(
    const float* __restrict__ logits, int* __restrict__ counts,
    int* __restrict__ eidx, int* __restrict__ slot, float* __restrict__ scale) {
  int t = blockIdx.x * blockDim.x + threadIdx.x;
  if (t >= T_TOK) return;
  const float* l = logits + (size_t)t * E_EXP;
  float best = l[0];
  int bi = 0;
#pragma unroll
  for (int e = 1; e < E_EXP; ++e) {
    float v = l[e];
    if (v > best) { best = v; bi = e; }
  }
  eidx[t]  = bi;
  scale[t] = fast_rcp(1.0f + __expf(-best));
  slot[t]  = atomicAdd(&counts[bi], 1);
}

// ============================================================
// 2) Exclusive prefix sum over 8 counts (trivial)
// ============================================================
__global__ void moe_offsets(const int* __restrict__ counts, int* __restrict__ offsets) {
  if (threadIdx.x == 0) {
    int s = 0;
#pragma unroll
    for (int e = 0; e < E_EXP; ++e) { offsets[e] = s; s += counts[e]; }
  }
}

// ============================================================
// 3) Scatter: xs_grouped[row] = bf16(x[t] * sigmoid(top_logit)), rowtok[row]=t
// ============================================================
__global__ __launch_bounds__(256) void moe_scatter(
    const float* __restrict__ x, const int* __restrict__ eidx,
    const int* __restrict__ slot, const float* __restrict__ scale,
    const int* __restrict__ offsets, unsigned short* __restrict__ xs,
    int* __restrict__ rowtok) {
  int t = blockIdx.x;
  int e = eidx[t];
  int row = offsets[e] + slot[t];
  float sc = scale[t];
  if (threadIdx.x == 0) rowtok[row] = t;
  unsigned int* dst = (unsigned int*)(xs + (size_t)row * H_DIM);
  const float2* src = (const float2*)(x + (size_t)t * H_DIM);
  for (int c = threadIdx.x; c < H_DIM / 2; c += 256) {
    float2 v = src[c];
    dst[c] = pkbf(v.x * sc, v.y * sc);
  }
}

// ============================================================
// 4) GEMM1: gate = xs @ w1[e]^T, up = xs @ w3[e]^T, h = silu(gate)*up (bf16)
//    w1/w3: [E, I, H] (K = H contiguous per output column)
//    A tile (16 x 64) double-buffered in LDS; one barrier per 64-K step;
//    4 independent WMMAs per barrier (split even/odd-K accumulators: no
//    intra-iteration WMMA->WMMA RAW hazard NOPs).
// ============================================================
__global__ __launch_bounds__(256) void moe_gemm1(
    const float* __restrict__ w1, const float* __restrict__ w3,
    const unsigned short* __restrict__ xs, const int* __restrict__ counts,
    const int* __restrict__ offsets, unsigned short* __restrict__ hbuf) {
  int e   = blockIdx.z;
  int cnt = counts[e];
  int mt  = blockIdx.y;
  if (mt * TM >= cnt) return;
  int rowBase = offsets[e] + mt * TM;
  int iBase   = blockIdx.x * TN;

  __shared__ __align__(16) unsigned short As[2][TM * BKT];  // 2 x 2KB

  int tid  = threadIdx.x;
  int wave = tid >> 5;
  int lane = tid & 31;
  int n    = lane & 15;
  int kOff = (lane < 16) ? 0 : 16;

  const float* pw1 = w1 + ((size_t)e * I_DIM + iBase + wave * 16 + n) * H_DIM + kOff;
  const float* pw3 = w3 + ((size_t)e * I_DIM + iBase + wave * 16 + n) * H_DIM + kOff;

  v8f accG0 = {}, accG1 = {};
  v8f accU0 = {}, accU1 = {};

  // A staging: thread t moves one uint2 (4 bf16); row = t>>4, col-pair = t&15
  int ar = tid >> 4, acp = tid & 15;
  const uint2* arowBase =
      (const uint2*)((const unsigned int*)xs + (size_t)(rowBase + ar) * (H_DIM / 2));

  ((uint2*)As[0])[tid] = arowBase[acp];
  __syncthreads();

  int p = 0;
  for (int k = 0; k < H_DIM; k += BKT, p ^= 1) {
    __builtin_prefetch(pw1 + k + 2 * BKT, 0, 3);
    __builtin_prefetch(pw3 + k + 2 * BKT, 0, 3);

    bool hasNext = (k + BKT < H_DIM);
    uint2 anext = {};
    if (hasNext) anext = arowBase[((k + BKT) >> 2) + acp];

    v16bf a0  = load_a_frag(As[p], lane, 0);
    v16bf a1  = load_a_frag(As[p], lane, 1);
    v16bf b10 = load_b_frag(pw1 + k);
    v16bf b30 = load_b_frag(pw3 + k);
    v16bf b11 = load_b_frag(pw1 + k + BK);
    v16bf b31 = load_b_frag(pw3 + k + BK);

    // all four are independent (distinct accumulators)
    accG0 = __builtin_amdgcn_wmma_f32_16x16x32_bf16(false, a0, false, b10,
                                                    (short)0, accG0, false, false);
    accU0 = __builtin_amdgcn_wmma_f32_16x16x32_bf16(false, a0, false, b30,
                                                    (short)0, accU0, false, false);
    accG1 = __builtin_amdgcn_wmma_f32_16x16x32_bf16(false, a1, false, b11,
                                                    (short)0, accG1, false, false);
    accU1 = __builtin_amdgcn_wmma_f32_16x16x32_bf16(false, a1, false, b31,
                                                    (short)0, accU1, false, false);

    if (hasNext) ((uint2*)As[p ^ 1])[tid] = anext;
    __syncthreads();
  }

  // Epilogue: C/D layout -> VGPR r holds M = r + (lane<16 ? 0 : 8), N = lane&15
  int mOff = (lane < 16) ? 0 : 8;
#pragma unroll
  for (int r = 0; r < 8; ++r) {
    int m = r + mOff;
    if (mt * TM + m < cnt) {
      float g = accG0[r] + accG1[r];
      float u = accU0[r] + accU1[r];
      float hv = g * fast_rcp(1.0f + __expf(-g)) * u;  // silu(g) * u
      hbuf[(size_t)(rowBase + m) * I_DIM + iBase + wave * 16 + n] = f2bf(hv);
    }
  }
}

// ============================================================
// 5) GEMM2: y = h @ w2[e]^T, scattered back to d_out[token]
//    w2: [E, H, I] (K = I contiguous per output column)
// ============================================================
__global__ __launch_bounds__(256) void moe_gemm2(
    const float* __restrict__ w2, const unsigned short* __restrict__ hbuf,
    const int* __restrict__ counts, const int* __restrict__ offsets,
    const int* __restrict__ rowtok, float* __restrict__ out) {
  int e   = blockIdx.z;
  int cnt = counts[e];
  int mt  = blockIdx.y;
  if (mt * TM >= cnt) return;
  int rowBase = offsets[e] + mt * TM;
  int hBase   = blockIdx.x * TN;

  __shared__ __align__(16) unsigned short As[2][TM * BKT];

  int tid  = threadIdx.x;
  int wave = tid >> 5;
  int lane = tid & 31;
  int n    = lane & 15;
  int kOff = (lane < 16) ? 0 : 16;

  const float* pw2 = w2 + ((size_t)e * H_DIM + hBase + wave * 16 + n) * I_DIM + kOff;

  v8f acc0 = {}, acc1 = {};
  int ar = tid >> 4, acp = tid & 15;
  const uint2* arowBase =
      (const uint2*)((const unsigned int*)hbuf + (size_t)(rowBase + ar) * (I_DIM / 2));

  ((uint2*)As[0])[tid] = arowBase[acp];
  __syncthreads();

  int p = 0;
  for (int k = 0; k < I_DIM; k += BKT, p ^= 1) {
    __builtin_prefetch(pw2 + k + 2 * BKT, 0, 3);

    bool hasNext = (k + BKT < I_DIM);
    uint2 anext = {};
    if (hasNext) anext = arowBase[((k + BKT) >> 2) + acp];

    v16bf a0 = load_a_frag(As[p], lane, 0);
    v16bf a1 = load_a_frag(As[p], lane, 1);
    v16bf b0 = load_b_frag(pw2 + k);
    v16bf b1 = load_b_frag(pw2 + k + BK);

    acc0 = __builtin_amdgcn_wmma_f32_16x16x32_bf16(false, a0, false, b0,
                                                   (short)0, acc0, false, false);
    acc1 = __builtin_amdgcn_wmma_f32_16x16x32_bf16(false, a1, false, b1,
                                                   (short)0, acc1, false, false);

    if (hasNext) ((uint2*)As[p ^ 1])[tid] = anext;
    __syncthreads();
  }

  int mOff = (lane < 16) ? 0 : 8;
#pragma unroll
  for (int r = 0; r < 8; ++r) {
    int m = r + mOff;
    if (mt * TM + m < cnt) {
      int t = rowtok[rowBase + m];
      out[(size_t)t * H_DIM + hBase + wave * 16 + n] = acc0[r] + acc1[r];
    }
  }
}

// ============================================================
extern "C" void kernel_launch(void* const* d_in, const int* in_sizes, int n_in,
                              void* d_out, int out_size, void* d_ws, size_t ws_size,
                              hipStream_t stream) {
  const float* x      = (const float*)d_in[0];
  const float* logits = (const float*)d_in[1];
  const float* w1     = (const float*)d_in[2];
  const float* w3     = (const float*)d_in[3];
  const float* w2     = (const float*)d_in[4];
  float* out          = (float*)d_out;

  char* ws = (char*)d_ws;
  size_t off = 0;
  int* counts  = (int*)(ws + off); off += 256;
  int* offsets = (int*)(ws + off); off += 256;
  int* eidx    = (int*)(ws + off); off += sizeof(int) * T_TOK;
  int* slot    = (int*)(ws + off); off += sizeof(int) * T_TOK;
  float* scale = (float*)(ws + off); off += sizeof(float) * T_TOK;
  int* rowtok  = (int*)(ws + off); off += sizeof(int) * (T_TOK + PAD_ROWS);
  off = (off + 255) & ~(size_t)255;
  unsigned short* xs = (unsigned short*)(ws + off);
  off += (size_t)(T_TOK + PAD_ROWS) * H_DIM * sizeof(unsigned short);
  off = (off + 255) & ~(size_t)255;
  unsigned short* hb = (unsigned short*)(ws + off);
  // hb needs (T_TOK+PAD_ROWS) * I_DIM * 2 bytes; total ws use ~8 MB

  hipMemsetAsync(counts, 0, E_EXP * sizeof(int), stream);

  moe_router<<<(T_TOK + 255) / 256, 256, 0, stream>>>(logits, counts, eidx, slot, scale);
  moe_offsets<<<1, 32, 0, stream>>>(counts, offsets);
  moe_scatter<<<T_TOK, 256, 0, stream>>>(x, eidx, slot, scale, offsets, xs, rowtok);

  dim3 g1(I_DIM / TN, T_TOK / TM, E_EXP);  // (32, 32, 8)
  moe_gemm1<<<g1, 256, 0, stream>>>(w1, w3, xs, counts, offsets, hb);

  dim3 g2(H_DIM / TN, T_TOK / TM, E_EXP);  // (16, 32, 8)
  moe_gemm2<<<g2, 256, 0, stream>>>(w2, hb, counts, offsets, rowtok, out);
}
